// ReadProcessEncoder_51015621541972
// MI455X (gfx1250) — compile-verified, matrix operand order checked by reading, not run on hardware
//
#include <hip/hip_runtime.h>
#include <hip/hip_bf16.h>

// CDNA5 (gfx1250) fused Read-Process-Write set encoder.
// One workgroup per batch row: WMMA f16 MLP -> memories in LDS ->
// 5 recurrent attention steps fully on-chip -> write MLP -> 64 floats out.

typedef __attribute__((ext_vector_type(16))) _Float16 v16h;
typedef __attribute__((ext_vector_type(8)))  float    v8f;

#define B_      4096
#define S_      512
#define E_      64
#define M_      16
#define H_      32
#define OUT_    64
#define STEPS_  5
#define NEG_INF_ (-1e30f)

__device__ __forceinline__ float sigmoidf_(float x) { return 1.0f / (1.0f + __expf(-x)); }

__device__ __forceinline__ v8f wmma_f16(v16h a, v16h b, v8f c) {
  // (neg_a, A, neg_b, B, c_mod, C, reuse_a, reuse_b)
  return __builtin_amdgcn_wmma_f32_16x16x32_f16(false, a, false, b, (short)0, c, false, false);
}

// Build a 32(K)x16(N) B-tile of W^T from row-major W (Nout x K), f32 -> f16.
// Lane l holds column n = nt*16 + (l&15); K mapping mirrors the A-matrix
// layout: lanes 0-15 -> K = {kb..kb+7, 16+kb..16+kb+7} with kb=0, lanes 16-31 kb=8.
__device__ __forceinline__ v16h make_b_tile(const float* __restrict__ W, int K,
                                            int nt, int kt, int lane) {
  int n  = nt * 16 + (lane & 15);
  int kb = (lane & 16) ? 8 : 0;
  const float* p0 = W + n * K + kt * 32 + kb;
  const float* p1 = p0 + 16;
  v16h b;
#pragma unroll
  for (int j = 0; j < 8; ++j) { b[j] = (_Float16)p0[j]; b[8 + j] = (_Float16)p1[j]; }
  return b;
}

// A-tile (16x32) from a per-lane global f32 row pointer (this lane's row).
__device__ __forceinline__ v16h make_a_global(const float* __restrict__ row, int kt, int lane) {
  int kb = (lane & 16) ? 8 : 0;
  const float* p0 = row + kt * 32 + kb;
  const float* p1 = p0 + 16;
  v16h a;
#pragma unroll
  for (int j = 0; j < 8; ++j) { a[j] = (_Float16)p0[j]; a[8 + j] = (_Float16)p1[j]; }
  return a;
}

// A-tile (16x32) from an LDS f16 staging tile with row stride 32.
__device__ __forceinline__ v16h make_a_lds(const _Float16* stg, int lane) {
  int r  = lane & 15;
  int kb = (lane & 16) ? 8 : 0;
  const _Float16* p0 = stg + r * 32 + kb;
  const _Float16* p1 = p0 + 16;
  v16h a;
#pragma unroll
  for (int j = 0; j < 8; ++j) { a[j] = p0[j]; a[8 + j] = p1[j]; }
  return a;
}

__global__ __launch_bounds__(256)
void rpe_fused_kernel(const float* __restrict__ input_set,
                      const int*   __restrict__ lengths,
                      const float* __restrict__ rw1, const float* __restrict__ rb1,
                      const float* __restrict__ rw2, const float* __restrict__ rb2,
                      const float* __restrict__ rw3, const float* __restrict__ rb3,
                      const float* __restrict__ whh,
                      const float* __restrict__ bih, const float* __restrict__ bhh,
                      const float* __restrict__ projw, const float* __restrict__ projb,
                      const float* __restrict__ ww1, const float* __restrict__ wb1,
                      const float* __restrict__ ww2, const float* __restrict__ wb2,
                      const float* __restrict__ ww3, const float* __restrict__ wb3,
                      const float* __restrict__ esv,
                      float* __restrict__ out)
{
  __shared__ __align__(16) _Float16 s_mem[S_ * M_];           // 16 KB: memories[b] f16
  __shared__ __align__(16) _Float16 s_stage[8 * 16 * 32];     // 8 KB: per-wave D->A bounce
  __shared__ float s_whh[4 * H_ * H_];                        // 16 KB
  __shared__ float s_gbias[4 * H_];
  __shared__ float s_proj[M_ * H_];
  __shared__ float s_projb[M_];
  __shared__ float s_gates[4 * H_];
  __shared__ float s_qstar[H_];
  __shared__ float s_c[H_], s_h[H_], s_q[M_];
  __shared__ float s_red[8];
  __shared__ float s_rw[8 * M_];
  __shared__ float s_bm, s_inv;
  __shared__ float s_tmp[64];

  const int tid  = threadIdx.x;
  const int lane = tid & 31;
  const int wave = tid >> 5;
  const int b    = blockIdx.x;

  // ------- cooperative loads of recurrent weights (used in phase 2) -------
  for (int i = tid; i < 4 * H_ * H_; i += 256) s_whh[i] = whh[i];
  for (int i = tid; i < 4 * H_;      i += 256) s_gbias[i] = bih[i] + bhh[i];
  for (int i = tid; i < M_ * H_;     i += 256) s_proj[i] = projw[i];
  if (tid < M_) s_projb[tid] = projb[tid];
  if (tid < H_) { s_qstar[tid] = 0.0f; s_c[tid] = 0.0f; }

  // ------- phase 1: memories = MLP(input_set[b]) via WMMA f16 -------
  v16h B1t[2][2], B2t[2], B3t;
#pragma unroll
  for (int kt = 0; kt < 2; ++kt)
#pragma unroll
    for (int nt = 0; nt < 2; ++nt)
      B1t[kt][nt] = make_b_tile(rw1, 64, nt, kt, lane);
  B2t[0] = make_b_tile(rw2, 32, 0, 0, lane);
  B2t[1] = make_b_tile(rw2, 32, 1, 0, lane);
  B3t    = make_b_tile(rw3, 32, 0, 0, lane);

  const float rb1lo = rb1[lane & 15], rb1hi = rb1[16 + (lane & 15)];
  const float rb2lo = rb2[lane & 15], rb2hi = rb2[16 + (lane & 15)];
  const float rb3lo = rb3[lane & 15];

  _Float16* stg = s_stage + wave * (16 * 32);
  const int n_  = lane & 15;
  const int mb_ = (lane & 16) ? 8 : 0;

  for (int t = wave; t < S_ / 16; t += 8) {
    const int s0 = t * 16;
    const float* xrow = input_set + ((size_t)b * S_ + s0 + (lane & 15)) * E_;

    // layer 1: (16x64) @ (64x32)
    v16h A0 = make_a_global(xrow, 0, lane);
    v16h A1 = make_a_global(xrow, 1, lane);
    v8f c0 = {}, c1 = {};
    c0 = wmma_f16(A0, B1t[0][0], c0);
    c0 = wmma_f16(A1, B1t[1][0], c0);
    c1 = wmma_f16(A0, B1t[0][1], c1);
    c1 = wmma_f16(A1, B1t[1][1], c1);
#pragma unroll
    for (int v = 0; v < 8; ++v) {
      stg[(mb_ + v) * 32 + n_]      = (_Float16)fmaxf(c0[v] + rb1lo, 0.0f);
      stg[(mb_ + v) * 32 + n_ + 16] = (_Float16)fmaxf(c1[v] + rb1hi, 0.0f);
    }
    // layer 2: (16x32) @ (32x32)   (wave-local LDS ops are in-order)
    v16h Ah = make_a_lds(stg, lane);
    v8f d0 = {}, d1 = {};
    d0 = wmma_f16(Ah, B2t[0], d0);
    d1 = wmma_f16(Ah, B2t[1], d1);
#pragma unroll
    for (int v = 0; v < 8; ++v) {
      stg[(mb_ + v) * 32 + n_]      = (_Float16)fmaxf(d0[v] + rb2lo, 0.0f);
      stg[(mb_ + v) * 32 + n_ + 16] = (_Float16)fmaxf(d1[v] + rb2hi, 0.0f);
    }
    // layer 3: (16x32) @ (32x16) -> memories tile
    v16h Ah2 = make_a_lds(stg, lane);
    v8f m0 = {};
    m0 = wmma_f16(Ah2, B3t, m0);
#pragma unroll
    for (int v = 0; v < 8; ++v)
      s_mem[(s0 + mb_ + v) * M_ + n_] = (_Float16)(m0[v] + rb3lo);
  }
  __syncthreads();

  // ------- phase 2: 5 recurrent attention steps, all on-chip -------
  const int len      = lengths[b];
  const int safe_len = (len > 0) ? len : 1;
  const bool validA  = (tid < safe_len);
  const bool validB  = (tid + 256 < safe_len);

  // each thread caches its two memory rows (s = tid, tid+256) in registers
  float mrow0[M_], mrow1[M_];
#pragma unroll
  for (int d = 0; d < M_; ++d) {
    mrow0[d] = (float)s_mem[(size_t)tid * M_ + d];
    mrow1[d] = (float)s_mem[(size_t)(tid + 256) * M_ + d];
  }

  for (int step = 0; step < STEPS_; ++step) {
    __syncthreads();
    // gates = q_star @ Whh^T + (bih + bhh)
    if (tid < 4 * H_) {
      float g = s_gbias[tid];
      const float* wr = &s_whh[tid * H_];
#pragma unroll
      for (int k = 0; k < H_; ++k) g += wr[k] * s_qstar[k];
      s_gates[tid] = g;
    }
    __syncthreads();
    // LSTM cell update (gate order i, f, g, o)
    if (tid < H_) {
      float ig = sigmoidf_(s_gates[tid]);
      float fg = sigmoidf_(s_gates[H_ + tid]);
      float gg = tanhf(s_gates[2 * H_ + tid]);
      float og = sigmoidf_(s_gates[3 * H_ + tid]);
      float cc = fg * s_c[tid] + ig * gg;
      s_c[tid] = cc;
      s_h[tid] = og * tanhf(cc);
    }
    __syncthreads();
    // q = h @ proj_w^T + proj_b
    if (tid < M_) {
      float q = s_projb[tid];
      const float* pr = &s_proj[tid * H_];
#pragma unroll
      for (int k = 0; k < H_; ++k) q += pr[k] * s_h[k];
      s_q[tid]     = q;
      s_qstar[tid] = q;
    }
    __syncthreads();

    // attention logits for s = tid and s = tid + 256
    float qv[M_];
#pragma unroll
    for (int d = 0; d < M_; ++d) qv[d] = s_q[d];
    float e0 = 0.0f, e1 = 0.0f;
#pragma unroll
    for (int d = 0; d < M_; ++d) { e0 += mrow0[d] * qv[d]; e1 += mrow1[d] * qv[d]; }
    if (!validA) e0 = NEG_INF_;
    if (!validB) e1 = NEG_INF_;

    // block max (wave32 shuffle + LDS)
    float mx = fmaxf(e0, e1);
#pragma unroll
    for (int off = 16; off > 0; off >>= 1) mx = fmaxf(mx, __shfl_xor(mx, off, 32));
    if (lane == 0) s_red[wave] = mx;
    __syncthreads();
    if (tid == 0) {
      float m = s_red[0];
      for (int w = 1; w < 8; ++w) m = fmaxf(m, s_red[w]);
      s_bm = m;
    }
    __syncthreads();
    const float bm = s_bm;
    float a0 = __expf(e0 - bm);
    float a1 = __expf(e1 - bm);

    // block sum
    float sm = a0 + a1;
#pragma unroll
    for (int off = 16; off > 0; off >>= 1) sm += __shfl_xor(sm, off, 32);
    if (lane == 0) s_red[wave] = sm;
    __syncthreads();
    if (tid == 0) {
      float s = 0.0f;
      for (int w = 0; w < 8; ++w) s += s_red[w];
      s_inv = 1.0f / s;
    }

    // r = sum_s a_s * memories[s]  (unnormalized partials, normalize at end)
    float pr[M_];
#pragma unroll
    for (int d = 0; d < M_; ++d) pr[d] = a0 * mrow0[d] + a1 * mrow1[d];
#pragma unroll
    for (int d = 0; d < M_; ++d) {
      float v = pr[d];
#pragma unroll
      for (int off = 16; off > 0; off >>= 1) v += __shfl_xor(v, off, 32);
      if (lane == 0) s_rw[wave * M_ + d] = v;
    }
    __syncthreads();
    if (tid < M_) {
      float r = 0.0f;
      for (int w = 0; w < 8; ++w) r += s_rw[w * M_ + tid];
      s_qstar[M_ + tid] = r * s_inv;
    }
  }
  __syncthreads();

  // ------- phase 3: write MLP 32 -> 32 -> 32 -> 64 -------
  if (tid < 32) {
    float h = wb1[tid];
    const float* wr = ww1 + tid * H_;
#pragma unroll
    for (int k = 0; k < H_; ++k) h += wr[k] * s_qstar[k];
    s_tmp[tid] = fmaxf(h, 0.0f);
  }
  __syncthreads();
  if (tid < 32) {
    float h = wb2[tid];
    const float* wr = ww2 + tid * 32;
#pragma unroll
    for (int k = 0; k < 32; ++k) h += wr[k] * s_tmp[k];
    s_tmp[32 + tid] = fmaxf(h, 0.0f);
  }
  __syncthreads();
  if (tid < OUT_) {
    float o = wb3[tid];
    const float* wr = ww3 + tid * 32;
#pragma unroll
    for (int k = 0; k < 32; ++k) o += wr[k] * s_tmp[32 + k];
    if (len == 0) o = esv[tid];
    out[(size_t)b * OUT_ + tid] = o;
  }
}

extern "C" void kernel_launch(void* const* d_in, const int* in_sizes, int n_in,
                              void* d_out, int out_size, void* d_ws, size_t ws_size,
                              hipStream_t stream) {
  (void)in_sizes; (void)n_in; (void)out_size; (void)d_ws; (void)ws_size;
  const float* input_set = (const float*)d_in[0];
  const int*   lengths   = (const int*)  d_in[1];
  const float* rw1  = (const float*)d_in[2];
  const float* rb1  = (const float*)d_in[3];
  const float* rw2  = (const float*)d_in[4];
  const float* rb2  = (const float*)d_in[5];
  const float* rw3  = (const float*)d_in[6];
  const float* rb3  = (const float*)d_in[7];
  // d_in[8] = lstm_wih: multiplies the all-zero LSTM input -> unused
  const float* whh  = (const float*)d_in[9];
  const float* bih  = (const float*)d_in[10];
  const float* bhh  = (const float*)d_in[11];
  const float* pjw  = (const float*)d_in[12];
  const float* pjb  = (const float*)d_in[13];
  const float* ww1  = (const float*)d_in[14];
  const float* wb1  = (const float*)d_in[15];
  const float* ww2  = (const float*)d_in[16];
  const float* wb2  = (const float*)d_in[17];
  const float* ww3  = (const float*)d_in[18];
  const float* wb3  = (const float*)d_in[19];
  const float* esv  = (const float*)d_in[20];
  float* out = (float*)d_out;

  rpe_fused_kernel<<<B_, 256, 0, stream>>>(input_set, lengths,
                                           rw1, rb1, rw2, rb2, rw3, rb3,
                                           whh, bih, bhh, pjw, pjb,
                                           ww1, wb1, ww2, wb2, ww3, wb3,
                                           esv, out);
}
